// NewLLMPhase2_69836168233292
// MI455X (gfx1250) — matrix operand order repaired and checked by reading, not compiled, not run on hardware
//
#include <hip/hip_runtime.h>
#include <hip/hip_bf16.h>

// ---------------------------------------------------------------------------
// Sizes (fixed by the reference)
// ---------------------------------------------------------------------------
#define VOCAB 32000
#define EMBD  512
#define CTXD  512
#define HIDD  1024
#define NBLK  4
#define BB    4
#define SS    256

typedef __attribute__((ext_vector_type(16))) __bf16 v16bf;
typedef __attribute__((ext_vector_type(8)))  __bf16 v8bf;
typedef __attribute__((ext_vector_type(8)))  float  v8f;

// Build a 16x32 bf16 A-fragment for v_wmma_f32_16x16x32_bf16 from a row-major
// source: lane l holds row (l&15); K values = kb..kb+7 and kb+16..kb+23 where
// kb = (l>>4)*8 (per CDNA5 ISA 7.12.2 "16-bit A-Matrix 16x32").
// Caller passes p = &row[(kstep*32) + (l>>4)*8]; two contiguous 16B chunks.
__device__ __forceinline__ v16bf make_afrag(const __bf16* p) {
  v8bf lo = *(const v8bf*)(p);
  v8bf hi = *(const v8bf*)(p + 16);
  return __builtin_shufflevector(lo, hi, 0,1,2,3,4,5,6,7,8,9,10,11,12,13,14,15);
}

// B-fragment load from packed layout [ntile][kchunk16][col16][k16]:
// two 16-byte loads per lane: k = 8h..8h+7 (chunk 2kk) and 16+8h..16+8h+7 (chunk 2kk+1).
__device__ __forceinline__ v16bf bfrag(const __bf16* base, int vt, int kk,
                                       int lhi, int l15) {
  const __bf16* p = base + ((size_t)(vt * 64 + kk * 2)) * 256 + l15 * 16 + lhi * 8;
  v8bf lo = *(const v8bf*)(p);
  v8bf hi = *(const v8bf*)(p + 256);
  return __builtin_shufflevector(lo, hi, 0,1,2,3,4,5,6,7,8,9,10,11,12,13,14,15);
}
// Same but K = 512 packed tables are identical layout (kchunks=64 only differs
// by vt stride); head GEMM uses 32 kchunks per vt:
__device__ __forceinline__ v16bf bfrag32(const __bf16* base, int vt, int kk,
                                         int lhi, int l15) {
  const __bf16* p = base + ((size_t)vt * 32 + kk * 2) * 256 + l15 * 16 + lhi * 8;
  v8bf lo = *(const v8bf*)(p);
  v8bf hi = *(const v8bf*)(p + 256);
  return __builtin_shufflevector(lo, hi, 0,1,2,3,4,5,6,7,8,9,10,11,12,13,14,15);
}

// ---------------------------------------------------------------------------
// Kernel 1: embedding gather + LayerNorm -> f32 x[B*S][EMBD]
// ---------------------------------------------------------------------------
__global__ __launch_bounds__(256) void embed_ln_kernel(
    const int* __restrict__ ids, const float* __restrict__ table,
    const float* __restrict__ gamma, const float* __restrict__ beta,
    float* __restrict__ xF) {
  __shared__ float red[16];
  const int row = blockIdx.x;            // b*SS + s
  const int tid = threadIdx.x;
  const int id  = ids[row];
  const float* e = table + (size_t)id * EMBD;
  float v0 = e[tid];
  float v1 = e[tid + 256];
  float s = v0 + v1;
  float q = v0 * v0 + v1 * v1;
#pragma unroll
  for (int o = 16; o > 0; o >>= 1) {
    s += __shfl_down(s, o, 32);
    q += __shfl_down(q, o, 32);
  }
  const int wave = tid >> 5, lane = tid & 31;
  if (lane == 0) { red[wave] = s; red[8 + wave] = q; }
  __syncthreads();
  if (tid == 0) {
    float ts = 0.f, tq = 0.f;
#pragma unroll
    for (int i = 0; i < 8; ++i) { ts += red[i]; tq += red[8 + i]; }
    red[0] = ts; red[8] = tq;
  }
  __syncthreads();
  const float mean = red[0] * (1.0f / EMBD);
  const float var  = red[8] * (1.0f / EMBD) - mean * mean;
  const float inv  = rsqrtf(var + 1e-5f);
  float* o = xF + (size_t)row * EMBD;
  o[tid]       = (v0 - mean) * inv * gamma[tid]       + beta[tid];
  o[tid + 256] = (v1 - mean) * inv * gamma[tid + 256] + beta[tid + 256];
}

// ---------------------------------------------------------------------------
// Kernel 2: pack f32 weights [NB][K][N] (row-major) into the WMMA B-fragment
// layout: bf16 [NB][N/16][K/16][col16][k16] -- each lane's 16-K B-frag slice
// is read as two 16-byte loads from adjacent k16 chunks.
// ---------------------------------------------------------------------------
__global__ __launch_bounds__(256) void pack_b_kernel(
    const float* __restrict__ src, __bf16* __restrict__ dst,
    int NB, int K, int N) {
  size_t idx = (size_t)blockIdx.x * blockDim.x + threadIdx.x;
  size_t total = (size_t)NB * K * N;
  if (idx >= total) return;
  const int ki = (int)(idx & 15);
  const int ci = (int)((idx >> 4) & 15);
  size_t r = idx >> 8;
  const int KC = K >> 4;
  const int kc = (int)(r % KC); r /= KC;
  const int NT = N >> 4;
  const int vt = (int)(r % NT);
  const int nb = (int)(r / NT);
  const float v = src[((size_t)nb * K + kc * 16 + ki) * N + vt * 16 + ci];
  dst[idx] = (__bf16)v;
}

// ---------------------------------------------------------------------------
// Kernel 3: sequential recurrent scan. One workgroup, 1024 threads = 32 waves.
// All matmuls are M=16-padded WMMAs (rows 0..3 = batch; rows 4..15 zero in z,
// so padded rows never pollute real rows). z = [ctx | tok] and h live in LDS
// as bf16; f32 ctx/tok masters live in registers of owning lanes.
// K-loops are double-buffered and share one A-frag across two WMMAs so loads
// overlap matrix ops instead of draining counters before every WMMA.
// ---------------------------------------------------------------------------
__global__ __launch_bounds__(1024) void scan_kernel(
    const float* __restrict__ xF,        // [BB][SS][EMBD] f32
    const __bf16* __restrict__ WhP,      // [4][64][64][256]
    const __bf16* __restrict__ WcP,      // [4][32][64][256]
    const __bf16* __restrict__ WtP,      // [4][32][64][256]
    const float* __restrict__ bh,        // [4][HIDD]
    const float* __restrict__ bc,        // [4][CTXD]
    const float* __restrict__ bt,        // [4][EMBD]
    __bf16* __restrict__ Aout)           // [4][1024][512] bf16 ctx rows
{
  __shared__ __attribute__((aligned(32))) __bf16 zB[16][HIDD];  // 32 KB
  __shared__ __attribute__((aligned(32))) __bf16 hB[16][HIDD];  // 32 KB

  const int tid  = threadIdx.x;
  const int wave = tid >> 5;      // 0..31
  const int lane = tid & 31;
  const int l15  = lane & 15;
  const int lhi  = lane >> 4;
  const int colw = wave * 16 + l15;   // this lane's owned column (0..511)

  // zero LDS (rows 4..15 of z must stay zero forever)
  for (int i = tid; i < 16 * HIDD; i += 1024) {
    ((__bf16*)zB)[i] = (__bf16)0.0f;
    ((__bf16*)hB)[i] = (__bf16)0.0f;
  }
  // f32 masters: cr[r] = ctx[b=r][colw], tr[r] = tok[b=r][colw] (lanes 0..15)
  float cr[4] = {0.f, 0.f, 0.f, 0.f};
  float tr[4];
  __syncthreads();

  for (int s = 0; s < SS; ++s) {
    // fresh tok from x[:, s, :]
    if (lhi == 0) {
#pragma unroll
      for (int r = 0; r < 4; ++r) {
        float v = xF[((size_t)r * SS + s) * EMBD + colw];
        tr[r] = v;
        zB[r][CTXD + colw] = (__bf16)v;
      }
    }
    __syncthreads();

    for (int blk = 0; blk < NBLK; ++blk) {
      // ---- phase 1: h = tanh(z @ Wh[blk] + bh[blk]) -------------------
      // Two N-tiles per wave share one A-fragment; double-buffered K-loop.
      {
        const __bf16* WhB = WhP + (size_t)blk * 64 * 64 * 256;
        const int vt0 = wave * 2, vt1 = vt0 + 1;
        v8f acc0 = {}, acc1 = {};
        v16bf a[2], b0[2], b1[2];
        a[0]  = make_afrag(&zB[l15][lhi * 8]);
        b0[0] = bfrag(WhB, vt0, 0, lhi, l15);
        b1[0] = bfrag(WhB, vt1, 0, lhi, l15);
#pragma unroll 4
        for (int kk = 0; kk < 32; ++kk) {      // K = 1024
          const int cur = kk & 1, nxt = cur ^ 1;
          if (kk + 1 < 32) {
            a[nxt]  = make_afrag(&zB[l15][(kk + 1) * 32 + lhi * 8]);
            b0[nxt] = bfrag(WhB, vt0, kk + 1, lhi, l15);
            b1[nxt] = bfrag(WhB, vt1, kk + 1, lhi, l15);
          }
          acc0 = __builtin_amdgcn_wmma_f32_16x16x32_bf16(
              false, a[cur], false, b0[cur], (short)0, acc0, false, false);
          acc1 = __builtin_amdgcn_wmma_f32_16x16x32_bf16(
              false, a[cur], false, b1[cur], (short)0, acc1, false, false);
        }
        const int col0 = vt0 * 16 + l15;
        const int col1 = vt1 * 16 + l15;
        const float bias0 = bh[blk * HIDD + col0];
        const float bias1 = bh[blk * HIDD + col1];
#pragma unroll
        for (int r = 0; r < 8; ++r) {
          hB[r + lhi * 8][col0] = (__bf16)tanhf(acc0[r] + bias0);
          hB[r + lhi * 8][col1] = (__bf16)tanhf(acc1[r] + bias1);
        }
      }
      __syncthreads();

      // ---- phase 2 (fused): ctx += h@Wc + bc ; tok += h@Wt + bt -------
      // Both outputs share the same A-fragments from h.
      {
        const __bf16* WcB = WcP + (size_t)blk * 32 * 64 * 256;
        const __bf16* WtB = WtP + (size_t)blk * 32 * 64 * 256;
        const int vt = wave;                   // 0..31 N-tile
        v8f accC = {}, accT = {};
        v16bf a[2], bcf[2], btf[2];
        a[0]   = make_afrag(&hB[l15][lhi * 8]);
        bcf[0] = bfrag(WcB, vt, 0, lhi, l15);
        btf[0] = bfrag(WtB, vt, 0, lhi, l15);
#pragma unroll 4
        for (int kk = 0; kk < 32; ++kk) {      // K = 1024
          const int cur = kk & 1, nxt = cur ^ 1;
          if (kk + 1 < 32) {
            a[nxt]   = make_afrag(&hB[l15][(kk + 1) * 32 + lhi * 8]);
            bcf[nxt] = bfrag(WcB, vt, kk + 1, lhi, l15);
            btf[nxt] = bfrag(WtB, vt, kk + 1, lhi, l15);
          }
          accC = __builtin_amdgcn_wmma_f32_16x16x32_bf16(
              false, a[cur], false, bcf[cur], (short)0, accC, false, false);
          accT = __builtin_amdgcn_wmma_f32_16x16x32_bf16(
              false, a[cur], false, btf[cur], (short)0, accT, false, false);
        }
        if (lhi == 0) {
          const float biasC = bc[blk * CTXD + colw];
          const float biasT = bt[blk * EMBD + colw];
#pragma unroll
          for (int r = 0; r < 4; ++r) {
            float nc = cr[r] + accC[r] + biasC;
            cr[r] = nc;
            zB[r][colw] = (__bf16)nc;
            Aout[((size_t)blk * 1024 + (size_t)r * SS + s) * CTXD + colw] = (__bf16)nc;
            float nt = tr[r] + accT[r] + biasT;
            tr[r] = nt;
            zB[r][CTXD + colw] = (__bf16)nt;
          }
        }
      }
      __syncthreads();
    } // blk
  } // s
}

// ---------------------------------------------------------------------------
// Kernel 4: vocab head GEMM: out[blk] = A[blk] @ Wout[blk] + bout[blk]
// A: [1024 x 512] bf16, W packed: [2000][32][256] bf16, out f32 [1024 x 32000].
// WG = 256 threads = 8 waves (4 along M x 2 along N), tile 128x128,
// each wave: 2 (M) x 4 (N) accumulator tiles, double-buffered K-loop of 16.
// ---------------------------------------------------------------------------
__global__ __launch_bounds__(256) void head_gemm_kernel(
    const __bf16* __restrict__ A,        // [4][1024][512]
    const __bf16* __restrict__ WP,       // [4][2000][32][256]
    const float* __restrict__ bout,      // [4][VOCAB]
    float* __restrict__ out)             // [4][1024][VOCAB]
{
  const int blk = blockIdx.z;
  const int m0w = blockIdx.y * 128;
  const int n0w = blockIdx.x * 128;
  const int tid = threadIdx.x;
  const int wave = tid >> 5;
  const int lane = tid & 31;
  const int l15  = lane & 15;
  const int lhi  = lane >> 4;
  const int wm = wave >> 1;             // 0..3
  const int wn = wave & 1;              // 0..1
  const int m0 = m0w + wm * 32;
  const int n0 = n0w + wn * 64;
  const int vt0 = n0 >> 4;

  const __bf16* Ab = A  + (size_t)blk * 1024 * 512;
  const __bf16* Wb = WP + (size_t)blk * 2000 * 32 * 256;

  v8f acc[2][4] = {};
  v16bf afr[2][2];
  v16bf bfr[2][4];

  // preload K-step 0
#pragma unroll
  for (int i = 0; i < 2; ++i) {
    afr[0][i] = make_afrag(Ab + (size_t)(m0 + i * 16 + l15) * 512 + lhi * 8);
  }
#pragma unroll
  for (int j = 0; j < 4; ++j) {
    bfr[0][j] = bfrag32(Wb, vt0 + j, 0, lhi, l15);
  }

#pragma unroll 2
  for (int kk = 0; kk < 16; ++kk) {     // K = 512
    const int cur = kk & 1, nxt = cur ^ 1;
    if (kk + 1 < 16) {
#pragma unroll
      for (int i = 0; i < 2; ++i) {
        afr[nxt][i] = make_afrag(Ab + (size_t)(m0 + i * 16 + l15) * 512 +
                                 (kk + 1) * 32 + lhi * 8);
      }
#pragma unroll
      for (int j = 0; j < 4; ++j) {
        bfr[nxt][j] = bfrag32(Wb, vt0 + j, kk + 1, lhi, l15);
      }
    }
#pragma unroll
    for (int j = 0; j < 4; ++j) {
#pragma unroll
      for (int i = 0; i < 2; ++i) {
        acc[i][j] = __builtin_amdgcn_wmma_f32_16x16x32_bf16(
            false, afr[cur][i], false, bfr[cur][j], (short)0, acc[i][j], false, false);
      }
    }
  }

#pragma unroll
  for (int j = 0; j < 4; ++j) {
    const int col = n0 + j * 16 + l15;
    const float bias = bout[(size_t)blk * VOCAB + col];
#pragma unroll
    for (int i = 0; i < 2; ++i) {
#pragma unroll
      for (int r = 0; r < 8; ++r) {
        const int row = m0 + i * 16 + r + lhi * 8;
        out[((size_t)blk * 1024 + row) * VOCAB + col] = acc[i][j][r] + bias;
      }
    }
  }
}

// ---------------------------------------------------------------------------
// Launch
// ---------------------------------------------------------------------------
extern "C" void kernel_launch(void* const* d_in, const int* in_sizes, int n_in,
                              void* d_out, int out_size, void* d_ws, size_t ws_size,
                              hipStream_t stream) {
  const int*   ids   = (const int*)d_in[0];
  const float* emb   = (const float*)d_in[1];
  const float* gamma = (const float*)d_in[2];
  const float* beta  = (const float*)d_in[3];
  const float* Wh    = (const float*)d_in[4];
  const float* bh    = (const float*)d_in[5];
  const float* Wc    = (const float*)d_in[6];
  const float* bc    = (const float*)d_in[7];
  const float* Wt    = (const float*)d_in[8];
  const float* bt    = (const float*)d_in[9];
  const float* Wout  = (const float*)d_in[10];
  const float* bout  = (const float*)d_in[11];
  float* out = (float*)d_out;

  char* ws = (char*)d_ws;
  size_t off = 0;
  auto wsalloc = [&](size_t bytes) -> char* {
    char* p = ws + off;
    off = (off + bytes + 255) & ~(size_t)255;
    return p;
  };

  float*  xF   = (float*) wsalloc((size_t)BB * SS * EMBD * sizeof(float));      // 2 MB
  __bf16* WhP  = (__bf16*)wsalloc((size_t)NBLK * HIDD * HIDD * 2);              // 8 MB
  __bf16* WcP  = (__bf16*)wsalloc((size_t)NBLK * HIDD * CTXD * 2);              // 4 MB
  __bf16* WtP  = (__bf16*)wsalloc((size_t)NBLK * HIDD * EMBD * 2);              // 4 MB
  __bf16* WoP  = (__bf16*)wsalloc((size_t)NBLK * CTXD * VOCAB * 2);             // 131 MB
  __bf16* Abuf = (__bf16*)wsalloc((size_t)NBLK * BB * SS * CTXD * 2);           // 4 MB

  // 1) embedding + layernorm
  embed_ln_kernel<<<BB * SS, 256, 0, stream>>>(ids, emb, gamma, beta, xF);

  // 2) pack weights into WMMA B-fragment layouts (bf16)
  {
    size_t tot;
    tot = (size_t)NBLK * (CTXD + EMBD) * HIDD;   // Wh: K=1024, N=1024
    pack_b_kernel<<<(unsigned)((tot + 255) / 256), 256, 0, stream>>>(Wh, WhP, NBLK, CTXD + EMBD, HIDD);
    tot = (size_t)NBLK * HIDD * CTXD;            // Wc: K=1024, N=512
    pack_b_kernel<<<(unsigned)((tot + 255) / 256), 256, 0, stream>>>(Wc, WcP, NBLK, HIDD, CTXD);
    tot = (size_t)NBLK * HIDD * EMBD;            // Wt: K=1024, N=512
    pack_b_kernel<<<(unsigned)((tot + 255) / 256), 256, 0, stream>>>(Wt, WtP, NBLK, HIDD, EMBD);
    tot = (size_t)NBLK * CTXD * VOCAB;           // Wout: K=512, N=32000
    pack_b_kernel<<<(unsigned)((tot + 255) / 256), 256, 0, stream>>>(Wout, WoP, NBLK, CTXD, VOCAB);
  }

  // 3) sequential scan (single workgroup, L2-resident weights, WMMA)
  scan_kernel<<<1, 1024, 0, stream>>>(xF, WhP, WcP, WtP, bh, bc, bt, Abuf);

  // 4) vocab head GEMM (dominant 134 GFLOP, bf16 WMMA, store-bandwidth bound)
  {
    dim3 grid(VOCAB / 128, (BB * SS) / 128, NBLK);  // 250 x 8 x 4
    head_gemm_kernel<<<grid, 256, 0, stream>>>(Abuf, WoP, bout, out);
  }
}
